// recurrent_GCLSTM_54202487275558
// MI455X (gfx1250) — matrix operand order repaired and checked by reading, not compiled, not run on hardware
//
#include <hip/hip_runtime.h>
#include <cstdint>
#include <cstddef>

typedef __attribute__((ext_vector_type(16))) __bf16 v16bf;
typedef __attribute__((ext_vector_type(8)))  float  v8f;

#define N_REAL 2800
#define NPAD   3000
#define F_IN   8192
#define H1     256
#define H2     128
#define NEDGE  179200

#define BLK_M 64
#define BLK_N 128
#define BLK_K 32
#define KPITCH (BLK_K + 8)   // 40 halves = 80 B row pitch: 16B-aligned, conflict-free

__device__ __forceinline__ void atomAddF(float* p, float v) { unsafeAtomicAdd(p, v); }

// ------------------------------------------------------------------
// Graph part: deg -> w_hat -> scatter T1
// ------------------------------------------------------------------
__global__ void gclstm_deg_accum(const long long* __restrict__ ei,
                                 const float* __restrict__ ew,
                                 float* __restrict__ deg) {
  int e = blockIdx.x * blockDim.x + threadIdx.x;
  if (e < NEDGE) atomAddF(&deg[(int)ei[e]], ew[e]);
}

__global__ void gclstm_what(const long long* __restrict__ ei,
                            const float* __restrict__ ew,
                            const float* __restrict__ deg,
                            float* __restrict__ what) {
  int e = blockIdx.x * blockDim.x + threadIdx.x;
  if (e >= NEDGE) return;
  int r = (int)ei[e];
  int c = (int)ei[NEDGE + e];
  float dr = deg[r], dc = deg[c];
  float ir = dr > 0.f ? rsqrtf(dr) : 0.f;
  float ic = dc > 0.f ? rsqrtf(dc) : 0.f;
  what[e] = -ir * ew[e] * ic;
}

__global__ __launch_bounds__(128)
void gclstm_scatter_t1(const long long* __restrict__ ei,
                       const float* __restrict__ what,
                       const float* __restrict__ h0,
                       float* __restrict__ T1) {
  int e = blockIdx.x;
  int j = threadIdx.x;
  int r = (int)ei[e];
  int c = (int)ei[NEDGE + e];
  atomAddF(&T1[c * H2 + j], what[e] * h0[r * H2 + j]);
}

// ------------------------------------------------------------------
// f32 -> bf16 hi/lo split (for B operands of the WMMA GEMMs)
// ------------------------------------------------------------------
__global__ void gclstm_f32_to_hilo(const float* __restrict__ src,
                                   __bf16* __restrict__ hi,
                                   __bf16* __restrict__ lo, int n) {
  int i = blockIdx.x * blockDim.x + threadIdx.x;
  if (i >= n) return;
  float f = src[i];
  __bf16 h = (__bf16)f;
  hi[i] = h;
  lo[i] = (__bf16)(f - (float)h);
}

// Bbig (512x512): rows 0..255 = Wx[g], 256..383 = cw0[g], 384..511 = cw1[g];
// col = g*128 + o
__global__ void gclstm_build_bbig(const float* __restrict__ Wx,
                                  const float* __restrict__ cw0,
                                  const float* __restrict__ cw1,
                                  __bf16* __restrict__ hi,
                                  __bf16* __restrict__ lo) {
  int i = blockIdx.x * blockDim.x + threadIdx.x;
  if (i >= 512 * 512) return;
  int r = i >> 9;
  int c = i & 511;
  int g = c >> 7;
  int o = c & 127;
  float f;
  if (r < 256)      f = Wx [g * H1 * H2 + r * H2 + o];
  else if (r < 384) f = cw0[g * H2 * H2 + (r - 256) * H2 + o];
  else              f = cw1[g * H2 * H2 + (r - 384) * H2 + o];
  __bf16 h = (__bf16)f;
  hi[i] = h;
  lo[i] = (__bf16)(f - (float)h);
}

__global__ void gclstm_bias512(const float* __restrict__ bg,
                               const float* __restrict__ cb,
                               float* __restrict__ bias) {
  int i = blockIdx.x * blockDim.x + threadIdx.x;
  if (i < 512) bias[i] = bg[i] + cb[i];
}

__global__ void gclstm_pack_acat(const float* __restrict__ h0,
                                 const float* __restrict__ T1,
                                 float* __restrict__ Acat) {
  int i = blockIdx.x * blockDim.x + threadIdx.x;
  if (i >= NPAD * H2) return;
  int n = i >> 7;
  int j = i & 127;
  Acat[n * 512 + 256 + j] = h0[i];
  Acat[n * 512 + 384 + j] = T1[i];
}

// ------------------------------------------------------------------
// WMMA GEMM: C(MxN) = A(f32, MxK) @ B(bf16 hi/lo, KxN) + bias[, relu]
// bf16x3 emulation of f32: Ahi*Bhi + Ahi*Blo + Alo*Bhi, f32 accumulate.
// 256 thr = 8 waves; block tile 64x128; wave tile 16x64 (four 16x16 accs).
// A converted to bf16 hi/lo at LDS staging time; fragment build is pure
// ds_load_b128 for both operands.
// ------------------------------------------------------------------
union U8bf { uint4 q; __bf16 h[8]; };
union U4bf { uint2 q; __bf16 h[4]; };
union V16  { v16bf v; uint4 q[2]; };

template <bool RELU>
__global__ __launch_bounds__(256)
void gclstm_gemm_bf16x3(const float* __restrict__ A, int lda, int Mvalid, int Mstore,
                        const __bf16* __restrict__ Bhi, const __bf16* __restrict__ Blo,
                        int N, int K, const float* __restrict__ bias,
                        float* __restrict__ C, int ldc) {
  __shared__ __bf16 Ash[BLK_M][KPITCH];   // A hi, [m][k]
  __shared__ __bf16 Asl[BLK_M][KPITCH];   // A lo
  __shared__ __bf16 Bsh[BLK_N][KPITCH];   // B hi, [n][k] (transposed)
  __shared__ __bf16 Bsl[BLK_N][KPITCH];   // B lo

  const int tid  = threadIdx.x;
  const int lane = tid & 31;
  const int wave = tid >> 5;
  const int wm   = wave & 3;   // 0..3 -> M sub-tile (16 rows)
  const int wn   = wave >> 2;  // 0..1 -> N sub-tile (64 cols)
  const int m0   = blockIdx.y * BLK_M;
  const int n0   = blockIdx.x * BLK_N;

  v8f acc[4] = {};

  // per-lane fragment coordinates
  const int am  = lane & 15;          // A row within 16
  const int akb = (lane >> 4) * 8;    // A K base: 0 or 8
  const int bn  = lane & 15;          // B col within 16
  const int bkb = (lane >> 4) * 8;    // B K base: 0 or 8

  // A staging coords: 64x32 f32, 8 elems/thread
  const int ar = tid >> 2;            // 0..63
  const int ac = (tid & 3) * 8;       // 0/8/16/24
  // B staging coords: threads 0..127 -> hi, 128..255 -> lo; 4(k)x8(n) each
  const int half = tid >> 7;
  const int bt   = tid & 127;
  const int bkr  = (bt >> 4) * 4;     // k base 0..28
  const int bnc  = (bt & 15) * 8;     // n base 0..120

  const __bf16* Bsrc = half ? Blo : Bhi;
  __bf16 (*BsD)[KPITCH] = half ? Bsl : Bsh;

  for (int k0 = 0; k0 < K; k0 += BLK_K) {
    // ---- stage A: load f32, split to bf16 hi/lo, store both to LDS ----
    {
      int gr = m0 + ar;
      float4 v0 = {0.f, 0.f, 0.f, 0.f}, v1 = {0.f, 0.f, 0.f, 0.f};
      if (gr < Mvalid) {
        const float* src = A + (size_t)gr * lda + k0 + ac;
        v0 = *(const float4*)src;
        v1 = *(const float4*)(src + 4);
        if (k0 + BLK_K < K)
          __builtin_prefetch(src + BLK_K, 0, 0);
      }
      float fs[8] = {v0.x, v0.y, v0.z, v0.w, v1.x, v1.y, v1.z, v1.w};
      U8bf uh, ul;
#pragma unroll
      for (int i = 0; i < 8; ++i) {
        __bf16 h = (__bf16)fs[i];
        uh.h[i] = h;
        ul.h[i] = (__bf16)(fs[i] - (float)h);
      }
      *(uint4*)&Ash[ar][ac] = uh.q;
      *(uint4*)&Asl[ar][ac] = ul.q;
    }
    // ---- stage B: 4(k)x8(n) per thread, store transposed [n][k] ----
    {
      const size_t boff = (size_t)(k0 + bkr) * N + n0 + bnc;
      U8bf r0, r1, r2, r3;
      r0.q = *(const uint4*)(Bsrc + boff);
      r1.q = *(const uint4*)(Bsrc + boff + N);
      r2.q = *(const uint4*)(Bsrc + boff + 2 * (size_t)N);
      r3.q = *(const uint4*)(Bsrc + boff + 3 * (size_t)N);
#pragma unroll
      for (int i = 0; i < 8; ++i) {
        U4bf cpk;
        cpk.h[0] = r0.h[i];
        cpk.h[1] = r1.h[i];
        cpk.h[2] = r2.h[i];
        cpk.h[3] = r3.h[i];
        *(uint2*)&BsD[bnc + i][bkr] = cpk.q;
      }
    }
    __syncthreads();

    // ---- fragments: pure ds_load_b128 ----
    V16 ahi, alo;
    {
      const int m = wm * 16 + am;
      ahi.q[0] = *(const uint4*)&Ash[m][akb];
      ahi.q[1] = *(const uint4*)&Ash[m][16 + akb];
      alo.q[0] = *(const uint4*)&Asl[m][akb];
      alo.q[1] = *(const uint4*)&Asl[m][16 + akb];
    }
#pragma unroll
    for (int s = 0; s < 4; ++s) {
      const int n = wn * 64 + s * 16 + bn;
      V16 bh, bl;
      bh.q[0] = *(const uint4*)&Bsh[n][bkb];
      bh.q[1] = *(const uint4*)&Bsh[n][bkb + 16];
      bl.q[0] = *(const uint4*)&Bsl[n][bkb];
      bl.q[1] = *(const uint4*)&Bsl[n][bkb + 16];
      acc[s] = __builtin_amdgcn_wmma_f32_16x16x32_bf16(false, ahi.v, false, bh.v, (short)0, acc[s], false, false);
      acc[s] = __builtin_amdgcn_wmma_f32_16x16x32_bf16(false, ahi.v, false, bl.v, (short)0, acc[s], false, false);
      acc[s] = __builtin_amdgcn_wmma_f32_16x16x32_bf16(false, alo.v, false, bh.v, (short)0, acc[s], false, false);
    }
    __syncthreads();
  }

  // ---- epilogue: C layout — VGPR r: lanes0-15 M=r, lanes16-31 M=8+r; N=lane&15
  const int rowb = m0 + wm * 16 + (lane >> 4) * 8;
  const int colb = n0 + wn * 64 + (lane & 15);
#pragma unroll
  for (int r = 0; r < 8; ++r) {
    int row = rowb + r;
    if (row < Mstore) {
#pragma unroll
      for (int s = 0; s < 4; ++s) {
        int col = colb + s * 16;
        float v = acc[s][r] + bias[col];
        if (RELU) v = fmaxf(v, 0.f);
        C[(size_t)row * ldc + col] = v;
      }
    }
  }
}

// ------------------------------------------------------------------
// LSTM pointwise + output projection
// ------------------------------------------------------------------
__global__ void gclstm_lstm_pw(const float* __restrict__ G,
                               const float* __restrict__ c0,
                               float* __restrict__ hnew) {
  int i = blockIdx.x * blockDim.x + threadIdx.x;
  if (i >= NPAD * H2) return;
  int n = i >> 7, j = i & 127;
  const float* g = G + (size_t)n * 512;
  float gi = g[j], gf = g[128 + j], gt = g[256 + j], go = g[384 + j];
  float iv = 1.f / (1.f + __expf(-gi));
  float fv = 1.f / (1.f + __expf(-gf));
  float tv = tanhf(gt);
  float ov = 1.f / (1.f + __expf(-go));
  float c  = fv * c0[i] + iv * tv;
  hnew[i]  = ov * tanhf(c);
}

__global__ __launch_bounds__(128)
void gclstm_out_proj(const float* __restrict__ h,
                     const float* __restrict__ fc_w,
                     const float* __restrict__ fc_b,
                     float* __restrict__ out) {
  __shared__ float red[128];
  int n = blockIdx.x;
  int j = threadIdx.x;
  red[j] = fmaxf(h[(size_t)n * H2 + j], 0.f) * fc_w[j];
  __syncthreads();
#pragma unroll
  for (int s = 64; s > 0; s >>= 1) {
    if (j < s) red[j] += red[j + s];
    __syncthreads();
  }
  if (j == 0) out[n] = red[0] + fc_b[0];
}

// ------------------------------------------------------------------
extern "C" void kernel_launch(void* const* d_in, const int* in_sizes, int n_in,
                              void* d_out, int out_size, void* d_ws, size_t ws_size,
                              hipStream_t stream) {
  const float*     x      = (const float*)d_in[0];
  const long long* ei     = (const long long*)d_in[1];
  const float*     ew     = (const float*)d_in[2];
  const float*     h0     = (const float*)d_in[3];
  const float*     c0     = (const float*)d_in[4];
  const float*     fc0_w  = (const float*)d_in[5];
  const float*     fc0_b  = (const float*)d_in[6];
  const float*     Wx     = (const float*)d_in[7];
  const float*     bg     = (const float*)d_in[8];
  const float*     cw0    = (const float*)d_in[9];
  const float*     cw1    = (const float*)d_in[10];
  const float*     cb     = (const float*)d_in[11];
  const float*     fc_w   = (const float*)d_in[12];
  const float*     fc_b   = (const float*)d_in[13];

  float* out  = (float*)d_out;
  float* hnew = out + N_REAL;  // 3000*128 f32

  // workspace layout
  char*  base = (char*)d_ws;
  size_t off  = 0;
  auto alloc = [&](size_t bytes) {
    void* p = base + off;
    off = (off + bytes + 255) & ~(size_t)255;
    return p;
  };
  __bf16* Bhi0   = (__bf16*)alloc((size_t)F_IN * H1 * 2);   // 4 MB
  __bf16* Blo0   = (__bf16*)alloc((size_t)F_IN * H1 * 2);   // 4 MB
  float*  Acat   = (float*) alloc((size_t)NPAD * 512 * 4);  // 6 MB
  __bf16* BbigHi = (__bf16*)alloc((size_t)512 * 512 * 2);
  __bf16* BbigLo = (__bf16*)alloc((size_t)512 * 512 * 2);
  float*  G      = (float*) alloc((size_t)NPAD * 512 * 4);  // 6 MB
  float*  T1     = (float*) alloc((size_t)NPAD * H2 * 4);
  float*  deg    = (float*) alloc((size_t)NPAD * 4);
  float*  what   = (float*) alloc((size_t)NEDGE * 4);
  float*  bias512= (float*) alloc(512 * 4);

  hipMemsetAsync(deg, 0, (size_t)NPAD * 4, stream);
  hipMemsetAsync(T1, 0, (size_t)NPAD * H2 * 4, stream);

  // graph pipeline
  gclstm_deg_accum<<<(NEDGE + 255) / 256, 256, 0, stream>>>(ei, ew, deg);
  gclstm_what<<<(NEDGE + 255) / 256, 256, 0, stream>>>(ei, ew, deg, what);
  gclstm_scatter_t1<<<NEDGE, 128, 0, stream>>>(ei, what, h0, T1);

  // B operand conversions
  gclstm_f32_to_hilo<<<(F_IN * H1 + 255) / 256, 256, 0, stream>>>(fc0_w, Bhi0, Blo0, F_IN * H1);
  gclstm_build_bbig<<<(512 * 512 + 255) / 256, 256, 0, stream>>>(Wx, cw0, cw1, BbigHi, BbigLo);
  gclstm_bias512<<<2, 256, 0, stream>>>(bg, cb, bias512);

  // z = relu(xp @ fc0_w + fc0_b) -> Acat[:, 0:256]
  {
    dim3 grid(H1 / BLK_N, (NPAD + BLK_M - 1) / BLK_M);
    gclstm_gemm_bf16x3<true><<<grid, 256, 0, stream>>>(
        x, F_IN, N_REAL, NPAD, Bhi0, Blo0, H1, F_IN, fc0_b, Acat, 512);
  }
  gclstm_pack_acat<<<(NPAD * H2 + 255) / 256, 256, 0, stream>>>(h0, T1, Acat);

  // gates = Acat(3000x512) @ Bbig(512x512) + (bg+cb)
  {
    dim3 grid(512 / BLK_N, (NPAD + BLK_M - 1) / BLK_M);
    gclstm_gemm_bf16x3<false><<<grid, 256, 0, stream>>>(
        Acat, 512, NPAD, NPAD, BbigHi, BbigLo, 512, 512, bias512, G, 512);
  }

  gclstm_lstm_pw<<<(NPAD * H2 + 255) / 256, 256, 0, stream>>>(G, c0, hnew);
  gclstm_out_proj<<<N_REAL, 128, 0, stream>>>(hnew, fc_w, fc_b, out);

  (void)in_sizes; (void)n_in; (void)out_size; (void)ws_size;
}
